// BaseX2HAttLayer_cross_87548613362076
// MI455X (gfx1250) — matrix operand order repaired and checked by reading, not compiled, not run on hardware
//
#include <hip/hip_runtime.h>

// ---------------------------------------------------------------------------
// MI455X (gfx1250) graph-attention layer.
// All GEMM work routed through v_wmma_f32_16x16x32_bf16 (wave32, 16x16 tiles).
// ---------------------------------------------------------------------------

typedef __bf16 bf16_t;
typedef bf16_t v16bf __attribute__((ext_vector_type(16)));
typedef bf16_t v8bf  __attribute__((ext_vector_type(8)));
typedef float  v8f   __attribute__((ext_vector_type(8)));
typedef float  v4f   __attribute__((ext_vector_type(4)));

#define LN_EPS 1e-5f

static __device__ __forceinline__ v8f wmma_bf16f32(v16bf a, v16bf b, v8f c) {
  // D = A(16x32 bf16) * B(32x16 bf16) + C(16x16 f32)
  return __builtin_amdgcn_wmma_f32_16x16x32_bf16(false, a, false, b, (short)0, c,
                                                 false, false);
}

static __device__ __forceinline__ v8f vzero8() {
  v8f z = {0.f, 0.f, 0.f, 0.f, 0.f, 0.f, 0.f, 0.f};
  return z;
}

// A-fragment (16x32 bf16) from an LDS row-major tile.
// ISA layout: lanes 0-15 hold row M=lane, K = 0..7 & 16..23; lanes 16-31 hold
// K = 8..15 & 24..31.
static __device__ __forceinline__ v16bf load_a_lds(const bf16_t* base, int stride,
                                                   int k0, int p, int half) {
  const bf16_t* row = base + p * stride;
  v8bf x = *(const v8bf*)(row + k0 + half * 8);
  v8bf y = *(const v8bf*)(row + k0 + 16 + half * 8);
  v16bf r;
#pragma unroll
  for (int i = 0; i < 8; ++i) { r[i] = x[i]; r[8 + i] = y[i]; }
  return r;
}

// A-fragment converting f32 global data on the fly (two 8-float chunks).
static __device__ __forceinline__ v16bf load_a_f32(const float* c0, const float* c1) {
  v16bf r;
#pragma unroll
  for (int i = 0; i < 8; ++i) { r[i] = (bf16_t)c0[i]; r[8 + i] = (bf16_t)c1[i]; }
  return r;
}

// B-fragment from the pre-swizzled bf16 weight image: fragment (kt,nt) is
// 32 lanes x 16 bf16 contiguous (one 32B load per lane).
static __device__ __forceinline__ v16bf load_b(const bf16_t* wswz, int ntiles,
                                               int kt, int nt, int lane) {
  return *(const v16bf*)(wswz + ((size_t)((kt * ntiles + nt) * 32 + lane)) * 16);
}

static __device__ __forceinline__ void add_bias8(v8f acc[8], const float* b, int p) {
#pragma unroll
  for (int t = 0; t < 8; ++t) {
    float bv = b[t * 16 + p];
#pragma unroll
    for (int r = 0; r < 8; ++r) acc[t][r] += bv;
  }
}

// Fused LayerNorm (over 128 hidden) + ReLU on a 16x128 C-layout tile.
// C layout: acc[t][r] = element(row = half*8 + r, col = t*16 + p).
static __device__ __forceinline__ void ln_relu(v8f acc[8], const float* g,
                                               const float* be, int p) {
  float s[8], ss[8];
#pragma unroll
  for (int r = 0; r < 8; ++r) { s[r] = 0.f; ss[r] = 0.f; }
#pragma unroll
  for (int t = 0; t < 8; ++t)
#pragma unroll
    for (int r = 0; r < 8; ++r) { float v = acc[t][r]; s[r] += v; ss[r] += v * v; }
#pragma unroll
  for (int m = 1; m < 16; m <<= 1) {
#pragma unroll
    for (int r = 0; r < 8; ++r) {
      s[r]  += __shfl_xor(s[r],  m, 16);
      ss[r] += __shfl_xor(ss[r], m, 16);
    }
  }
  float mu[8], rs[8];
#pragma unroll
  for (int r = 0; r < 8; ++r) {
    mu[r] = s[r] * (1.f / 128.f);
    float var = ss[r] * (1.f / 128.f) - mu[r] * mu[r];
    rs[r] = rsqrtf(var + LN_EPS);
  }
#pragma unroll
  for (int t = 0; t < 8; ++t) {
    float gv = g[t * 16 + p], bv = be[t * 16 + p];
#pragma unroll
    for (int r = 0; r < 8; ++r) {
      float h = (acc[t][r] - mu[r]) * rs[r] * gv + bv;
      acc[t][r] = fmaxf(h, 0.f);
    }
  }
}

// Re-stage a C-layout f32 tile as a row-major bf16 16x128 LDS tile (A for stage 2).
static __device__ __forceinline__ void stage_c_bf16(bf16_t* stg, const v8f acc[8],
                                                    int p, int half) {
#pragma unroll
  for (int t = 0; t < 8; ++t)
#pragma unroll
    for (int r = 0; r < 8; ++r)
      stg[(half * 8 + r) * 128 + t * 16 + p] = (bf16_t)acc[t][r];
}

// 16xDIN (LDS, bf16) @ DINx128 (swizzled) -> 16x128 accumulate.
template <int DIN>
static __device__ __forceinline__ void gemm_lds_x8(v8f acc[8], const bf16_t* A,
                                                   const bf16_t* Wswz, int p,
                                                   int half, int lane) {
#pragma unroll 1
  for (int kt = 0; kt < DIN / 32; ++kt) {
    v16bf a = load_a_lds(A, DIN, kt * 32, p, half);
#pragma unroll
    for (int nt = 0; nt < 8; ++nt)
      acc[nt] = wmma_bf16f32(a, load_b(Wswz, 8, kt, nt, lane), acc[nt]);
  }
}

// Monotonic float<->uint encoding for atomicMax-based segment max.
static __device__ __forceinline__ unsigned fenc(float f) {
  unsigned u = __float_as_uint(f);
  return (u & 0x80000000u) ? ~u : (u | 0x80000000u);
}
static __device__ __forceinline__ float fdec(unsigned u) {
  return (u & 0x80000000u) ? __uint_as_float(u & 0x7fffffffu) : __uint_as_float(~u);
}

// ---------------------------------------------------------------------------
// Weight pre-swizzle: f32 row-major (din x dout) -> bf16 WMMA-B fragment image.
// dst[(((kt*ntiles)+nt)*32 + lane)*16 + i] = W[(kt*32 + (i/8)*16 + (lane/16)*8
//                                              + i%8)*dout + nt*16 + (lane%16)]
// ---------------------------------------------------------------------------
__global__ void swz_kernel(const float* __restrict__ W, bf16_t* __restrict__ out,
                           int din, int dout) {
  int idx = blockIdx.x * 256 + threadIdx.x;
  int total = din * dout;
  if (idx >= total) return;
  int i = idx & 15;
  int lane = (idx >> 4) & 31;
  int frag = idx >> 9;
  int ntiles = dout >> 4;
  int nt = frag % ntiles;
  int kt = frag / ntiles;
  int half = lane >> 4;
  int p = lane & 15;
  int k = kt * 32 + (i >> 3) * 16 + half * 8 + (i & 7);
  int n = nt * 16 + p;
  out[idx] = (bf16_t)W[(size_t)k * dout + n];
}

__global__ void init_kernel(float* agg, float* outx, float* denom,
                            unsigned* segmax, int N) {
  int t = blockIdx.x * 256 + threadIdx.x;
  if (t < N * 128) agg[t] = 0.f;
  if (t < N * 48) outx[t] = 0.f;
  if (t < N * 16) { denom[t] = 0.f; segmax[t] = 0u; }  // 0u == encoded -inf
}

// ---------------------------------------------------------------------------
// q = hq-MLP(h_in): per-wave 16-node tile, WMMA GEMMs with fused LN/ReLU.
// ---------------------------------------------------------------------------
__global__ __launch_bounds__(128) void q_kernel(
    const float* __restrict__ h_in, const bf16_t* __restrict__ W1s,
    const bf16_t* __restrict__ W2s, const float* __restrict__ b1,
    const float* __restrict__ g, const float* __restrict__ be,
    const float* __restrict__ b2, float* __restrict__ qout, int N) {
  __shared__ alignas(32) bf16_t stg[4][16][128];
  __shared__ alignas(16) float stgf[4][16][128];
  const int wave = threadIdx.x >> 5, lane = threadIdx.x & 31;
  const int p = lane & 15, half = lane >> 4;
  const int n0 = (blockIdx.x * 4 + wave) * 16;
  const int row = min(n0 + p, N - 1);
  const float* rp = h_in + (size_t)row * 128;

  v8f acc[8];
  for (int t = 0; t < 8; ++t) acc[t] = vzero8();
#pragma unroll 1
  for (int kt = 0; kt < 4; ++kt) {
    v16bf a = load_a_f32(rp + kt * 32 + half * 8, rp + kt * 32 + 16 + half * 8);
#pragma unroll
    for (int nt = 0; nt < 8; ++nt)
      acc[nt] = wmma_bf16f32(a, load_b(W1s, 8, kt, nt, lane), acc[nt]);
  }
  add_bias8(acc, b1, p);
  ln_relu(acc, g, be, p);
  stage_c_bf16(&stg[wave][0][0], acc, p, half);
  __syncthreads();
  for (int t = 0; t < 8; ++t) acc[t] = vzero8();
  gemm_lds_x8<128>(acc, &stg[wave][0][0], W2s, p, half, lane);
  add_bias8(acc, b2, p);
#pragma unroll
  for (int t = 0; t < 8; ++t)
#pragma unroll
    for (int r = 0; r < 8; ++r)
      stgf[wave][half * 8 + r][t * 16 + p] = acc[t][r];
  __syncthreads();
  const v4f* sf = (const v4f*)&stgf[wave][0][0];
  v4f* qo = (v4f*)(qout + (size_t)n0 * 128);
  for (int c = lane; c < 512; c += 32)
    if (n0 + (c >> 5) < N) qo[c] = sf[c];
}

// ---------------------------------------------------------------------------
// Edge kernel: gather kv tile -> hk/hv/xv MLPs (WMMA) -> logits + atomicMax,
// bf16 v tile, xv tile.  4 waves/block, 16 edges/wave, E % 64 == 0.
// ---------------------------------------------------------------------------
__global__ __launch_bounds__(128) void edge_kernel(
    const float* __restrict__ h_in, const float* __restrict__ r_feat,
    const int* __restrict__ edge_index, const float* __restrict__ qbuf,
    const bf16_t* __restrict__ hkW1s, const bf16_t* __restrict__ hkW2s,
    const float* __restrict__ hk_b1, const float* __restrict__ hk_g,
    const float* __restrict__ hk_be, const float* __restrict__ hk_b2,
    const bf16_t* __restrict__ hvW1s, const bf16_t* __restrict__ hvW2s,
    const float* __restrict__ hv_b1, const float* __restrict__ hv_g,
    const float* __restrict__ hv_be, const float* __restrict__ hv_b2,
    const bf16_t* __restrict__ xvW1s, const bf16_t* __restrict__ xvW2s,
    const float* __restrict__ xv_b1, const float* __restrict__ xv_g,
    const float* __restrict__ xv_be, const float* __restrict__ xv_b2,
    float* __restrict__ logits, float* __restrict__ xvout,
    bf16_t* __restrict__ vout, unsigned* __restrict__ segmax, int E) {
  __shared__ alignas(32) bf16_t kvA[4][16][320];
  __shared__ alignas(32) bf16_t stg[4][16][128];
  __shared__ int dstI[4][16];

  const int wave = threadIdx.x >> 5;
  const int lane = threadIdx.x & 31;
  const int p = lane & 15;
  const int half = lane >> 4;
  const int e0 = (blockIdx.x * 4 + wave) * 16;

  // Gather kv = [r_feat | h_in[src] | h_in[dst]] (320 feats) into LDS as bf16.
  for (int m = 0; m < 16; ++m) {
    const int e = e0 + m;
    const int dn = edge_index[e];       // dst
    const int sn = edge_index[E + e];   // src
    for (int c = lane; c < 80; c += 32) {
      const float* sp;
      if (c < 16)      sp = r_feat + (size_t)e * 64 + c * 4;
      else if (c < 48) sp = h_in + (size_t)sn * 128 + (c - 16) * 4;
      else             sp = h_in + (size_t)dn * 128 + (c - 48) * 4;
      v4f x = *(const v4f*)sp;
      bf16_t* dp = &kvA[wave][m][c * 4];
      dp[0] = (bf16_t)x[0]; dp[1] = (bf16_t)x[1];
      dp[2] = (bf16_t)x[2]; dp[3] = (bf16_t)x[3];
    }
  }
  if (lane < 16) dstI[wave][lane] = edge_index[e0 + lane];
  __syncthreads();

  const bf16_t* A = &kvA[wave][0][0];
  bf16_t* S = &stg[wave][0][0];
  v8f acc[8];

  // ================= hk =================
  for (int t = 0; t < 8; ++t) acc[t] = vzero8();
  gemm_lds_x8<320>(acc, A, hkW1s, p, half, lane);
  add_bias8(acc, hk_b1, p);
  ln_relu(acc, hk_g, hk_be, p);
  stage_c_bf16(S, acc, p, half);
  __syncthreads();
  for (int t = 0; t < 8; ++t) acc[t] = vzero8();
  gemm_lds_x8<128>(acc, S, hkW2s, p, half, lane);
  add_bias8(acc, hk_b2, p);
  __syncthreads();
  stage_c_bf16(S, acc, p, half);  // k tile (bf16) in staging
  __syncthreads();

  // logits = (q[dst] . k) / sqrt(8), plus running segment max via atomicMax.
  {
    const float inv_sqrt_hd = 0.35355339059327373f;
#pragma unroll
    for (int j = 0; j < 8; ++j) {
      const int pi = lane + 32 * j;  // 256 (row,head) pairs per wave
      const int row = pi & 15;
      const int head = pi >> 4;
      const int e = e0 + row;
      const int d = dstI[wave][row];
      const bf16_t* kr = &stg[wave][row][head * 8];
      const float* qr = qbuf + (size_t)d * 128 + head * 8;
      float sum = 0.f;
#pragma unroll
      for (int t = 0; t < 8; ++t) sum += qr[t] * (float)kr[t];
      sum *= inv_sqrt_hd;
      logits[(size_t)e * 16 + head] = sum;
      atomicMax(&segmax[(size_t)d * 16 + head], fenc(sum));
    }
  }
  __syncthreads();

  // ================= hv =================
  for (int t = 0; t < 8; ++t) acc[t] = vzero8();
  gemm_lds_x8<320>(acc, A, hvW1s, p, half, lane);
  add_bias8(acc, hv_b1, p);
  ln_relu(acc, hv_g, hv_be, p);
  stage_c_bf16(S, acc, p, half);
  __syncthreads();
  for (int t = 0; t < 8; ++t) acc[t] = vzero8();
  gemm_lds_x8<128>(acc, S, hvW2s, p, half, lane);
  add_bias8(acc, hv_b2, p);
  __syncthreads();
  stage_c_bf16(S, acc, p, half);  // v tile (bf16) in staging
  __syncthreads();
  {
    v8bf* dd = (v8bf*)(vout + (size_t)e0 * 128);
    const v8bf* ssrc = (const v8bf*)S;
    for (int c = lane; c < 256; c += 32) dd[c] = ssrc[c];  // coalesced 16B chunks
  }
  __syncthreads();

  // ================= xv =================
  for (int t = 0; t < 8; ++t) acc[t] = vzero8();
  gemm_lds_x8<320>(acc, A, xvW1s, p, half, lane);
  add_bias8(acc, xv_b1, p);
  ln_relu(acc, xv_g, xv_be, p);
  stage_c_bf16(S, acc, p, half);
  __syncthreads();
  v8f a1 = vzero8();
#pragma unroll
  for (int kt = 0; kt < 4; ++kt) {
    v16bf af = load_a_lds(S, 128, kt * 32, p, half);
    a1 = wmma_bf16f32(af, load_b(xvW2s, 1, kt, 0, lane), a1);
  }
  {
    float bv = xv_b2[p];
#pragma unroll
    for (int r = 0; r < 8; ++r)
      xvout[(size_t)(e0 + half * 8 + r) * 16 + p] = a1[r] + bv;
  }
}

// exp(logit - segmax[dst]) and segment-sum denominator.
__global__ void softmax_denom_kernel(const int* __restrict__ edge_index,
                                     float* __restrict__ logits,
                                     const unsigned* __restrict__ segmax,
                                     float* __restrict__ denom, int E) {
  int t = blockIdx.x * 256 + threadIdx.x;
  if (t >= E * 16) return;
  int e = t >> 4, h = t & 15;
  int d = edge_index[e];
  float m = fdec(segmax[(size_t)d * 16 + h]);
  float ex = __expf(logits[t] - m);
  logits[t] = ex;
  atomicAdd(denom + (size_t)d * 16 + h, ex);
}

// Weighted scatter: agg[dst] += alpha * v ; outx[dst] += alpha * xv * rel_x.
__global__ __launch_bounds__(256) void scatter_kernel(
    const int* __restrict__ edge_index, const float* __restrict__ exbuf,
    const float* __restrict__ denom, const bf16_t* __restrict__ vout,
    const float* __restrict__ xvout, const float* __restrict__ rel_x,
    float* __restrict__ agg, float* __restrict__ outx, int E) {
  int t = threadIdx.x;
  int e0 = blockIdx.x * 2;
  int e = e0 + (t >> 7);
  int n = t & 127;
  int d = edge_index[e];
  int h = n >> 3;
  float alpha = exbuf[(size_t)e * 16 + h] / denom[(size_t)d * 16 + h];
  atomicAdd(agg + (size_t)d * 128 + n, alpha * (float)vout[(size_t)e * 128 + n]);
  if (t < 96) {
    int e2 = e0 + t / 48;
    int i = t % 48;
    int h2 = i / 3, c = i % 3;
    int d2 = edge_index[e2];
    float a2 = exbuf[(size_t)e2 * 16 + h2] / denom[(size_t)d2 * 16 + h2];
    atomicAdd(outx + (size_t)d2 * 48 + i,
              a2 * xvout[(size_t)e2 * 16 + h2] * rel_x[(size_t)e2 * 3 + c]);
  }
}

// node MLP over concat[agg, h_in] (din=256) + residual -> d_out[0 .. N*128).
__global__ __launch_bounds__(128) void node_kernel(
    const float* __restrict__ agg, const float* __restrict__ h_in,
    const bf16_t* __restrict__ W1s, const bf16_t* __restrict__ W2s,
    const float* __restrict__ b1, const float* __restrict__ g,
    const float* __restrict__ be, const float* __restrict__ b2,
    float* __restrict__ outp, int N) {
  __shared__ alignas(32) bf16_t stg[4][16][128];
  __shared__ alignas(16) float stgf[4][16][128];
  const int wave = threadIdx.x >> 5, lane = threadIdx.x & 31;
  const int p = lane & 15, half = lane >> 4;
  const int n0 = (blockIdx.x * 4 + wave) * 16;
  const int row = min(n0 + p, N - 1);
  const float* ra = agg + (size_t)row * 128;
  const float* rh = h_in + (size_t)row * 128;

  v8f acc[8];
  for (int t = 0; t < 8; ++t) acc[t] = vzero8();
#pragma unroll 1
  for (int kt = 0; kt < 8; ++kt) {
    const int k0 = kt * 32;
    const float* base = (k0 < 128) ? (ra + k0) : (rh + (k0 - 128));
    v16bf a = load_a_f32(base + half * 8, base + 16 + half * 8);
#pragma unroll
    for (int nt = 0; nt < 8; ++nt)
      acc[nt] = wmma_bf16f32(a, load_b(W1s, 8, kt, nt, lane), acc[nt]);
  }
  add_bias8(acc, b1, p);
  ln_relu(acc, g, be, p);
  stage_c_bf16(&stg[wave][0][0], acc, p, half);
  __syncthreads();
  for (int t = 0; t < 8; ++t) acc[t] = vzero8();
  gemm_lds_x8<128>(acc, &stg[wave][0][0], W2s, p, half, lane);
  add_bias8(acc, b2, p);
#pragma unroll
  for (int t = 0; t < 8; ++t)
#pragma unroll
    for (int r = 0; r < 8; ++r)
      stgf[wave][half * 8 + r][t * 16 + p] = acc[t][r];
  __syncthreads();
  const v4f* sf = (const v4f*)&stgf[wave][0][0];
  const v4f* hb = (const v4f*)(h_in + (size_t)n0 * 128);
  v4f* oo = (v4f*)(outp + (size_t)n0 * 128);
  for (int c = lane; c < 512; c += 32)
    if (n0 + (c >> 5) < N) { v4f v = sf[c] + hb[c]; oo[c] = v; }
}

// output_x = mean over heads -> d_out[N*128 .. N*128 + N*3).
__global__ void xmean_kernel(const float* __restrict__ outx,
                             float* __restrict__ out, int N) {
  int t = blockIdx.x * 256 + threadIdx.x;
  if (t >= N * 3) return;
  int n = t / 3, c = t % 3;
  float s = 0.f;
#pragma unroll
  for (int h = 0; h < 16; ++h) s += outx[(size_t)n * 48 + h * 3 + c];
  out[(size_t)N * 128 + t] = s * (1.f / 16.f);
}

// ---------------------------------------------------------------------------
extern "C" void kernel_launch(void* const* d_in, const int* in_sizes, int n_in,
                              void* d_out, int out_size, void* d_ws, size_t ws_size,
                              hipStream_t stream) {
  (void)n_in; (void)out_size; (void)ws_size;
  const float* h_in  = (const float*)d_in[0];
  const float* rel_x = (const float*)d_in[1];
  const float* r_feat = (const float*)d_in[3];
  const int* edge_index = (const int*)d_in[4];
  const int N = in_sizes[0] / 128;
  const int E = in_sizes[1] / 3;

  // params (flattened dict order): hk, hv, xv, hq, node; each W1,b1,g,be,W2,b2
  const float* hkW1 = (const float*)d_in[5];
  const float* hkb1 = (const float*)d_in[6];
  const float* hkg  = (const float*)d_in[7];
  const float* hkbe = (const float*)d_in[8];
  const float* hkW2 = (const float*)d_in[9];
  const float* hkb2 = (const float*)d_in[10];
  const float* hvW1 = (const float*)d_in[11];
  const float* hvb1 = (const float*)d_in[12];
  const float* hvg  = (const float*)d_in[13];
  const float* hvbe = (const float*)d_in[14];
  const float* hvW2 = (const float*)d_in[15];
  const float* hvb2 = (const float*)d_in[16];
  const float* xvW1 = (const float*)d_in[17];
  const float* xvb1 = (const float*)d_in[18];
  const float* xvg  = (const float*)d_in[19];
  const float* xvbe = (const float*)d_in[20];
  const float* xvW2 = (const float*)d_in[21];
  const float* xvb2 = (const float*)d_in[22];
  const float* hqW1 = (const float*)d_in[23];
  const float* hqb1 = (const float*)d_in[24];
  const float* hqg  = (const float*)d_in[25];
  const float* hqbe = (const float*)d_in[26];
  const float* hqW2 = (const float*)d_in[27];
  const float* hqb2 = (const float*)d_in[28];
  const float* ndW1 = (const float*)d_in[29];
  const float* ndb1 = (const float*)d_in[30];
  const float* ndg  = (const float*)d_in[31];
  const float* ndbe = (const float*)d_in[32];
  const float* ndW2 = (const float*)d_in[33];
  const float* ndb2 = (const float*)d_in[34];

  // workspace carve-out
  char* wsb = (char*)d_ws;
  size_t off = 0;
  auto carve = [&](size_t bytes) -> void* {
    void* pp = wsb + off;
    off += (bytes + 255) & ~(size_t)255;
    return pp;
  };
  bf16_t* hkW1s = (bf16_t*)carve((size_t)320 * 128 * 2);
  bf16_t* hkW2s = (bf16_t*)carve((size_t)128 * 128 * 2);
  bf16_t* hvW1s = (bf16_t*)carve((size_t)320 * 128 * 2);
  bf16_t* hvW2s = (bf16_t*)carve((size_t)128 * 128 * 2);
  bf16_t* xvW1s = (bf16_t*)carve((size_t)320 * 128 * 2);
  bf16_t* xvW2s = (bf16_t*)carve((size_t)128 * 16 * 2);
  bf16_t* hqW1s = (bf16_t*)carve((size_t)128 * 128 * 2);
  bf16_t* hqW2s = (bf16_t*)carve((size_t)128 * 128 * 2);
  bf16_t* ndW1s = (bf16_t*)carve((size_t)256 * 128 * 2);
  bf16_t* ndW2s = (bf16_t*)carve((size_t)128 * 128 * 2);
  float* qbuf   = (float*)carve((size_t)N * 128 * 4);
  float* logits = (float*)carve((size_t)E * 16 * 4);
  float* xvout  = (float*)carve((size_t)E * 16 * 4);
  bf16_t* vout  = (bf16_t*)carve((size_t)E * 128 * 2);
  unsigned* segmax = (unsigned*)carve((size_t)N * 16 * 4);
  float* denom  = (float*)carve((size_t)N * 16 * 4);
  float* agg    = (float*)carve((size_t)N * 128 * 4);
  float* outx   = (float*)carve((size_t)N * 48 * 4);

  auto swz = [&](const float* W, bf16_t* dst, int din, int dout) {
    int tot = din * dout;
    swz_kernel<<<(tot + 255) / 256, 256, 0, stream>>>(W, dst, din, dout);
  };
  swz(hkW1, hkW1s, 320, 128); swz(hkW2, hkW2s, 128, 128);
  swz(hvW1, hvW1s, 320, 128); swz(hvW2, hvW2s, 128, 128);
  swz(xvW1, xvW1s, 320, 128); swz(xvW2, xvW2s, 128, 16);
  swz(hqW1, hqW1s, 128, 128); swz(hqW2, hqW2s, 128, 128);
  swz(ndW1, ndW1s, 256, 128); swz(ndW2, ndW2s, 128, 128);

  init_kernel<<<(N * 128 + 255) / 256, 256, 0, stream>>>(agg, outx, denom, segmax, N);

  const int node_blocks = ((N + 15) / 16 + 3) / 4;
  q_kernel<<<node_blocks, 128, 0, stream>>>(h_in, hqW1s, hqW2s, hqb1, hqg, hqbe,
                                            hqb2, qbuf, N);

  edge_kernel<<<E / 64, 128, 0, stream>>>(
      h_in, r_feat, edge_index, qbuf,
      hkW1s, hkW2s, hkb1, hkg, hkbe, hkb2,
      hvW1s, hvW2s, hvb1, hvg, hvbe, hvb2,
      xvW1s, xvW2s, xvb1, xvg, xvbe, xvb2,
      logits, xvout, vout, segmax, E);

  softmax_denom_kernel<<<(E * 16 + 255) / 256, 256, 0, stream>>>(
      edge_index, logits, segmax, denom, E);

  scatter_kernel<<<E / 2, 256, 0, stream>>>(edge_index, logits, denom, vout,
                                            xvout, rel_x, agg, outx, E);

  node_kernel<<<node_blocks, 128, 0, stream>>>(agg, h_in, ndW1s, ndW2s, ndb1,
                                               ndg, ndbe, ndb2, (float*)d_out, N);

  xmean_kernel<<<(N * 3 + 255) / 256, 256, 0, stream>>>(outx, (float*)d_out, N);
}
